// SioConvPSBlock_13709535609452
// MI455X (gfx1250) — matrix operand order repaired and checked by reading, not compiled, not run on hardware
//
#include <hip/hip_runtime.h>
#include <hip/hip_bf16.h>
#include <math.h>

typedef __bf16 bf16;
typedef __bf16 v16bf __attribute__((ext_vector_type(16)));
typedef __bf16 v8bf  __attribute__((ext_vector_type(8)));
typedef float  v8f   __attribute__((ext_vector_type(8)));

#define D_    1024
#define FF_   4096
#define B_    4
#define L_    4096
#define NTOK  (B_ * L_)      // 16384
#define NCH   32             // chunks along L for the scan
#define CL    (L_ / NCH)     // 128
#define EPSF  1e-6f

// ---------------------------------------------------------------- helpers
__device__ __forceinline__ float softplusf(float x) {
    if (x > 20.f)  return x;
    if (x < -20.f) return __expf(x);
    return log1pf(__expf(x));
}
__device__ __forceinline__ float lsef(float a, float b) {  // log(e^a + e^b)
    float m = fmaxf(a, b), n = fminf(a, b);
    if (m == -INFINITY) return -INFINITY;
    return m + log1pf(__expf(n - m));
}

// ---------------------------------------------------------------- cvt f32 -> bf16
__global__ void cvt_f32_bf16(const float* __restrict__ s, bf16* __restrict__ d, int n) {
    int i = blockIdx.x * blockDim.x + threadIdx.x;
    if (i < n) d[i] = (bf16)s[i];
}

// ---------------------------------------------------------------- RMSNorm (f32 in, bf16 out)
__global__ void rmsnorm_bf16(const float* __restrict__ x, const float* __restrict__ g,
                             bf16* __restrict__ o) {
    __shared__ float red[256];
    int row = blockIdx.x;
    const float* xr = x + (size_t)row * D_;
    float v[4];
    float s = 0.f;
#pragma unroll
    for (int i = 0; i < 4; ++i) {
        v[i] = xr[threadIdx.x + i * 256];
        s += v[i] * v[i];
    }
    red[threadIdx.x] = s;
    __syncthreads();
    for (int off = 128; off > 0; off >>= 1) {
        if (threadIdx.x < off) red[threadIdx.x] += red[threadIdx.x + off];
        __syncthreads();
    }
    float scale = rsqrtf(red[0] * (1.f / (float)D_) + EPSF);
    bf16* orow = o + (size_t)row * D_;
#pragma unroll
    for (int i = 0; i < 4; ++i) {
        int d = threadIdx.x + i * 256;
        orow[d] = (bf16)(v[i] * scale * g[d]);
    }
}

// ---------------------------------------------------------------- WMMA bf16 GEMM
// C[rows, OUT] = A[rows, K](bf16) @ W[OUT, K]^T(bf16) + epilogue
// Block: 256 threads = 8 waves (4x2). Wave tile: 32x64 (2x4 of 16x16 WMMA).
// Block tile: 128x128. rows = 16384 (mult 128), OUT in {1024,4096}, K in {1024,4096}.
// Per wave per 32-K step: 8 WMMA for 12 b128 loads (1.5 loads/wmma).
union Frag { v16bf v; v8bf h[2]; };

template <int EPI>
__global__ __launch_bounds__(256) void gemm_wmma_bf16(
    const bf16* __restrict__ A, const bf16* __restrict__ W,
    const float* __restrict__ bias, const bf16* __restrict__ gate,
    const float* __restrict__ res, float* __restrict__ outF, bf16* __restrict__ outB,
    int K, int OUT)
{
    const int lane  = threadIdx.x & 31;
    const int wave  = threadIdx.x >> 5;
    const int wm    = wave >> 1;          // 0..3
    const int wn    = wave & 1;           // 0..1
    const int row0  = blockIdx.y * 128 + wm * 32;
    const int col0  = blockIdx.x * 128 + wn * 64;
    const int mlane = lane & 15;
    const int hi    = lane >> 4;          // 0 or 1 (lane half)

    v8f acc[2][4] = {};

    for (int k = 0; k < K; k += 32) {
        Frag af[2], bfr[4];
        // A 16x32 bf16 fragment: lanes 0-15 -> K {0..7,16..23}; lanes 16-31 -> +8
#pragma unroll
        for (int i = 0; i < 2; ++i) {
            const bf16* pa = A + (size_t)(row0 + i * 16 + mlane) * K + (k + hi * 8);
            af[i].h[0] = *(const v8bf*)(pa);
            af[i].h[1] = *(const v8bf*)(pa + 16);
        }
        // B 32x16 bf16 fragment: lanes 0-15 -> K {0..7,16..23}; lanes 16-31 -> +8
#pragma unroll
        for (int j = 0; j < 4; ++j) {
            const bf16* pb = W + (size_t)(col0 + j * 16 + mlane) * K + (k + hi * 8);
            bfr[j].h[0] = *(const v8bf*)(pb);
            bfr[j].h[1] = *(const v8bf*)(pb + 16);
        }
#pragma unroll
        for (int i = 0; i < 2; ++i)
#pragma unroll
            for (int j = 0; j < 4; ++j)
                acc[i][j] = __builtin_amdgcn_wmma_f32_16x16x32_bf16(
                    false, af[i].v, false, bfr[j].v, (short)0, acc[i][j], false, false);
    }

    // C/D layout: VGPR r -> row = r + 8*lane_half ; col = lane&15
#pragma unroll
    for (int i = 0; i < 2; ++i) {
#pragma unroll
        for (int j = 0; j < 4; ++j) {
            int col = col0 + j * 16 + mlane;
            float bcol = bias[col];
#pragma unroll
            for (int r = 0; r < 8; ++r) {
                int row = row0 + i * 16 + r + hi * 8;
                size_t idx = (size_t)row * OUT + col;
                float v = acc[i][j][r] + bcol;
                if (EPI == 0) {                    // plain f32
                    outF[idx] = v;
                } else if (EPI == 1) {             // silu -> bf16
                    outB[idx] = (bf16)(v / (1.f + __expf(-v)));
                } else if (EPI == 2) {             // gated + residual -> f32
                    outF[idx] = v * (float)gate[idx] + res[idx];
                } else if (EPI == 3) {             // gated -> bf16
                    outB[idx] = (bf16)(v * (float)gate[idx]);
                } else {                           // EPI == 4: + residual -> f32
                    outF[idx] = v + res[idx];
                }
            }
        }
    }
}

// ---------------------------------------------------------------- chunked scan
// Per (b,d): o_l = -softplus(dt); cum_l = cumsum(o); a_l = -softplus(-lnz) - softplus(-dt)
// lcse_l = logcumsumexp(a_l - cum_l); h_l = exp(lcse_l + cum_l) + exp(cum_l)*h0[d]
// Chunked: pass1 per-chunk totals, pass2 chunk prefixes, pass3 final values.

__global__ void scan_pass1(const float* __restrict__ lnz, const float* __restrict__ dt,
                           float* __restrict__ S, float* __restrict__ M) {
    int tid = blockIdx.x * blockDim.x + threadIdx.x;   // B*D*NCH threads
    int d = tid % D_;
    int c = (tid / D_) % NCH;
    int b = tid / (D_ * NCH);
    size_t base = ((size_t)(b * L_ + c * CL)) * D_ + d;
    float cs = 0.f, r = -INFINITY;
    for (int l = 0; l < CL; ++l) {
        size_t idx = base + (size_t)l * D_;
        float z = lnz[idx], t = dt[idx];
        float a = -softplusf(-z) - softplusf(-t);
        cs += -softplusf(t);
        r = lsef(r, a - cs);
    }
    size_t pi = (size_t)(b * NCH + c) * D_ + d;
    S[pi] = cs;
    M[pi] = r;
}

__global__ void scan_pass2(const float* __restrict__ S, const float* __restrict__ M,
                           float* __restrict__ Cprev, float* __restrict__ Lprev) {
    int tid = blockIdx.x * blockDim.x + threadIdx.x;   // B*D threads
    int d = tid % D_;
    int b = tid / D_;
    float C = 0.f, Lp = -INFINITY;
    for (int c = 0; c < NCH; ++c) {
        size_t pi = (size_t)(b * NCH + c) * D_ + d;
        Cprev[pi] = C;
        Lprev[pi] = Lp;
        Lp = lsef(Lp, M[pi] - C);
        C += S[pi];
    }
}

__global__ void scan_pass3(const float* __restrict__ lnz, const float* __restrict__ dt,
                           const float* __restrict__ Cprev, const float* __restrict__ Lprev,
                           const float* __restrict__ h0, bf16* __restrict__ h) {
    int tid = blockIdx.x * blockDim.x + threadIdx.x;   // B*D*NCH threads
    int d = tid % D_;
    int c = (tid / D_) % NCH;
    int b = tid / (D_ * NCH);
    size_t pi = (size_t)(b * NCH + c) * D_ + d;
    float Cp = Cprev[pi], Lp = Lprev[pi];
    float h0d = h0[d];
    size_t base = ((size_t)(b * L_ + c * CL)) * D_ + d;
    float cs = 0.f, r = -INFINITY;
    for (int l = 0; l < CL; ++l) {
        size_t idx = base + (size_t)l * D_;
        float z = lnz[idx], t = dt[idx];
        float a = -softplusf(-z) - softplusf(-t);
        cs += -softplusf(t);
        r = lsef(r, a - cs);
        float cum  = Cp + cs;
        float lcse = lsef(Lp, r - Cp);
        h[idx] = (bf16)(__expf(lcse + cum) + __expf(cum) * h0d);
    }
}

// ---------------------------------------------------------------- launch
extern "C" void kernel_launch(void* const* d_in, const int* in_sizes, int n_in,
                              void* d_out, int out_size, void* d_ws, size_t ws_size,
                              hipStream_t stream) {
    (void)in_sizes; (void)n_in; (void)out_size; (void)ws_size;
    const float* x       = (const float*)d_in[0];
    const float* w_ln_z  = (const float*)d_in[1];
    const float* b_ln_z  = (const float*)d_in[2];
    const float* w_dt    = (const float*)d_in[3];
    const float* b_dt    = (const float*)d_in[4];
    const float* w_y     = (const float*)d_in[5];
    const float* b_y     = (const float*)d_in[6];
    const float* w_yact  = (const float*)d_in[7];
    const float* b_yact  = (const float*)d_in[8];
    const float* h0      = (const float*)d_in[9];
    const float* w_fc    = (const float*)d_in[10];
    const float* b_fc    = (const float*)d_in[11];
    const float* w_fcact = (const float*)d_in[12];
    const float* b_fcact = (const float*)d_in[13];
    const float* w_out   = (const float*)d_in[14];
    const float* b_out   = (const float*)d_in[15];
    const float* g_sio   = (const float*)d_in[16];
    const float* g_ffn   = (const float*)d_in[17];

    char* ws = (char*)d_ws;
    size_t off = 0;
    auto alloc = [&](size_t bytes) -> void* {
        off = (off + 255) & ~(size_t)255;
        void* p = ws + off;
        off += bytes;
        return p;
    };

    bf16* wzB     = (bf16*)alloc((size_t)D_ * D_ * 2);
    bf16* wdtB    = (bf16*)alloc((size_t)D_ * D_ * 2);
    bf16* wyB     = (bf16*)alloc((size_t)D_ * D_ * 2);
    bf16* wyactB  = (bf16*)alloc((size_t)D_ * D_ * 2);
    bf16* wfcB    = (bf16*)alloc((size_t)FF_ * D_ * 2);
    bf16* wfcactB = (bf16*)alloc((size_t)FF_ * D_ * 2);
    bf16* woutB   = (bf16*)alloc((size_t)D_ * FF_ * 2);
    bf16* xn      = (bf16*)alloc((size_t)NTOK * D_ * 2);
    float* lnz    = (float*)alloc((size_t)NTOK * D_ * 4);
    float* dtb    = (float*)alloc((size_t)NTOK * D_ * 4);
    bf16* syact   = (bf16*)alloc((size_t)NTOK * D_ * 2);
    bf16* hbuf    = (bf16*)alloc((size_t)NTOK * D_ * 2);
    float* x1     = (float*)alloc((size_t)NTOK * D_ * 4);
    float* Sc     = (float*)alloc((size_t)B_ * D_ * NCH * 4);
    float* Mc     = (float*)alloc((size_t)B_ * D_ * NCH * 4);
    float* Cp     = (float*)alloc((size_t)B_ * D_ * NCH * 4);
    float* Lp     = (float*)alloc((size_t)B_ * D_ * NCH * 4);
    bf16* hid     = (bf16*)alloc((size_t)NTOK * FF_ * 2);
    // aliases over dead regions:
    bf16* xn2  = xn;            // xn dead after the first three GEMMs
    bf16* sact = (bf16*)lnz;    // lnz+dtb (128MB) dead after scan; sact needs exactly 128MB

    const int DD = D_ * D_, FD = FF_ * D_;
    cvt_f32_bf16<<<(DD + 255) / 256, 256, 0, stream>>>(w_ln_z,  wzB,     DD);
    cvt_f32_bf16<<<(DD + 255) / 256, 256, 0, stream>>>(w_dt,    wdtB,    DD);
    cvt_f32_bf16<<<(DD + 255) / 256, 256, 0, stream>>>(w_y,     wyB,     DD);
    cvt_f32_bf16<<<(DD + 255) / 256, 256, 0, stream>>>(w_yact,  wyactB,  DD);
    cvt_f32_bf16<<<(FD + 255) / 256, 256, 0, stream>>>(w_fc,    wfcB,    FD);
    cvt_f32_bf16<<<(FD + 255) / 256, 256, 0, stream>>>(w_fcact, wfcactB, FD);
    cvt_f32_bf16<<<(FD + 255) / 256, 256, 0, stream>>>(w_out,   woutB,   FD);

    // --- SioConvPS sublayer ---
    rmsnorm_bf16<<<NTOK, 256, 0, stream>>>(x, g_sio, xn);

    dim3 gD(D_ / 128, NTOK / 128);    // 8 x 128
    dim3 gF(FF_ / 128, NTOK / 128);   // 32 x 128
    gemm_wmma_bf16<0><<<gD, 256, 0, stream>>>(xn, wzB,    b_ln_z, nullptr, nullptr, lnz, nullptr, D_, D_);
    gemm_wmma_bf16<0><<<gD, 256, 0, stream>>>(xn, wdtB,   b_dt,   nullptr, nullptr, dtb, nullptr, D_, D_);
    gemm_wmma_bf16<1><<<gD, 256, 0, stream>>>(xn, wyactB, b_yact, nullptr, nullptr, nullptr, syact, D_, D_);

    scan_pass1<<<(B_ * D_ * NCH) / 256, 256, 0, stream>>>(lnz, dtb, Sc, Mc);
    scan_pass2<<<(B_ * D_) / 256, 256, 0, stream>>>(Sc, Mc, Cp, Lp);
    scan_pass3<<<(B_ * D_ * NCH) / 256, 256, 0, stream>>>(lnz, dtb, Cp, Lp, h0, hbuf);

    // x1 = x + (h @ w_y^T + b_y) * silu(yact)
    gemm_wmma_bf16<2><<<gD, 256, 0, stream>>>(hbuf, wyB, b_y, syact, x, x1, nullptr, D_, D_);

    // --- SwiGLU FFN sublayer ---
    rmsnorm_bf16<<<NTOK, 256, 0, stream>>>(x1, g_ffn, xn2);
    gemm_wmma_bf16<1><<<gF, 256, 0, stream>>>(xn2, wfcactB, b_fcact, nullptr, nullptr, nullptr, sact, D_, FF_);
    gemm_wmma_bf16<3><<<gF, 256, 0, stream>>>(xn2, wfcB,    b_fc,    sact,    nullptr, nullptr, hid,  D_, FF_);
    // out = x1 + hid @ w_out^T + b_out
    gemm_wmma_bf16<4><<<gD, 256, 0, stream>>>(hid, woutB, b_out, nullptr, x1, (float*)d_out, nullptr, FF_, D_);
}
